// KANLinear_49641232007351
// MI455X (gfx1250) — matrix operand order repaired and checked by reading, not compiled, not run on hardware
//
#include <hip/hip_runtime.h>

typedef __attribute__((ext_vector_type(16))) _Float16 v16h;
typedef __attribute__((ext_vector_type(8)))  _Float16 v8h;
typedef __attribute__((ext_vector_type(8)))  float    v8f;

#define IN_F    512
#define OUT_F   512
#define BATCH   16384
#define COEFF   8
#define K_TOT   (IN_F + IN_F * COEFF)   // 4608
#define BK      32
#define KTILES  (K_TOT / BK)            // 144
#define KSILU   (IN_F / BK)             // 16 K-tiles covering the silu(x) block
#define BM      128
#define BN      256
#define NTHR    512                     // 16 waves: 4 row-groups x 4 col-groups
#define PITCH   (BK + 8)                // 40 halves = 80B rows -> 16B-aligned b128 slots

// ---------------------------------------------------------------------------
// Weight prep: WcT[n][k] f16 (K-contiguous per output column), k in [0,4608):
//   k <  512 : base_weight[n][k]
//   k >= 512 : spline_weight[n][i][c]*spline_scaler[n][i], i=(k-512)/8, c=(k-512)%8
// All reads AND the write are coalesced along k. 4.5 MB in d_ws, L2-resident.
// ---------------------------------------------------------------------------
__global__ __launch_bounds__(256) void kan_prep_w(
    const float* __restrict__ bw, const float* __restrict__ sw,
    const float* __restrict__ ss, _Float16* __restrict__ WcT)
{
    int idx = blockIdx.x * 256 + threadIdx.x;
    if (idx >= OUT_F * K_TOT) return;
    int n = idx / K_TOT;
    int k = idx - n * K_TOT;
    float w;
    if (k < IN_F) {
        w = bw[n * IN_F + k];
    } else {
        int kp = k - IN_F;                       // = i*8 + c
        w = sw[n * (IN_F * COEFF) + kp] * ss[n * IN_F + (kp >> 3)];
    }
    WcT[idx] = (_Float16)w;
}

__device__ __forceinline__ float siluf(float v) {
    return v / (1.0f + __expf(-v));
}

// ---------------------------------------------------------------------------
// Fused KAN GEMM: out[16384,512] = [silu(x) | bspline(x)] (16384 x 4608)
//                                  @ WcT (4608 x 512), f16 in / f32 accum.
// 128x256 tile, 512 threads (16 waves, 4x4; each wave 32x64 via 8 WMMA/K-step).
// Double-buffered LDS; B tile filled with async global->LDS copies (ASYNCcnt),
// A tile (silu + closed-form cubic B-spline) recomputed in VALU per K-step.
// All 12 fragment ds_load_b128 issue before the 8-WMMA burst so the XDL pipe
// gets back-to-back WMMA issue instead of a dscnt stall every 2 WMMAs.
// ---------------------------------------------------------------------------
__global__ __launch_bounds__(NTHR) void kan_gemm(
    const float* __restrict__ x, const _Float16* __restrict__ WcT,
    float* __restrict__ out)
{
    __shared__ _Float16 sA[2][BM][PITCH];   // A tile, row-major (M x K)   20 KB
    __shared__ _Float16 sB[2][BN][PITCH];   // B tile, transposed (N x K)  40 KB

    const int tid  = threadIdx.x;
    const int lane = tid & 31;
    const int wave = tid >> 5;
    const int wm   = wave & 3;    // 0..3 : 32-row group
    const int wn   = wave >> 2;   // 0..3 : 64-col group
    const int half = lane >> 4;   // lane[4]
    const int l16  = lane & 15;

    const int rowBase = blockIdx.y * BM;
    const int colBase = blockIdx.x * BN;

    v8f acc[2][4];
#pragma unroll
    for (int am = 0; am < 2; ++am)
#pragma unroll
        for (int bn = 0; bn < 4; ++bn)
            acc[am][bn] = (v8f){0.f,0.f,0.f,0.f,0.f,0.f,0.f,0.f};

    // ---- A-tile fill: 128x32 f16 ----
    auto fillA = [&](int buf, int t) {
        int row = tid >> 2;                            // 0..127
        if (t < KSILU) {
            // silu(x): 4 threads per row, 8 contiguous cols each
            int cg = (tid & 3) * 8;
            const float4* gx =
                (const float4*)(x + (size_t)(rowBase + row) * IN_F + t * BK + cg);
            float4 v0 = gx[0], v1 = gx[1];
            v8h hv;
            hv[0] = (_Float16)siluf(v0.x); hv[1] = (_Float16)siluf(v0.y);
            hv[2] = (_Float16)siluf(v0.z); hv[3] = (_Float16)siluf(v0.w);
            hv[4] = (_Float16)siluf(v1.x); hv[5] = (_Float16)siluf(v1.y);
            hv[6] = (_Float16)siluf(v1.z); hv[7] = (_Float16)siluf(v1.w);
            *(v8h*)&sA[buf][row][cg] = hv;
        } else {
            // spline block: 32 K-slots = 4 features x 8 bases; 1 eval/thread.
            // Cubic B-spline on uniform knots g_j=-2.2+0.4j: only 4 of the 8
            // bases are nonzero; closed-form weights + zero-fill & scatter.
            int f  = tid & 3;
            int i0 = (t - KSILU) * 4;
            float xv = x[(size_t)(rowBase + row) * IN_F + i0 + f];
            float tp = (xv + 2.2f) * 2.5f;             // knot coordinate
            float fj = floorf(tp);
            fj = fminf(fmaxf(fj, 3.0f), 7.0f);         // x in [-1,1) -> j in [3,7]
            float u  = tp - fj;
            float u2 = u * u, u3 = u2 * u;
            const float s6 = 1.0f / 6.0f;
            float w0 = s6 * (1.f - 3.f*u + 3.f*u2 - u3);       // basis j-3
            float w1 = s6 * (4.f - 6.f*u2 + 3.f*u3);           // basis j-2
            float w2 = s6 * (1.f + 3.f*u + 3.f*u2 - 3.f*u3);   // basis j-1
            float w3 = s6 * u3;                                 // basis j
            _Float16* dst = &sA[buf][row][f * 8];
            *(v8h*)dst = (v8h){0,0,0,0,0,0,0,0};
            int b0 = (int)fj - 3;                      // 0..4
            dst[b0 + 0] = (_Float16)w0;
            dst[b0 + 1] = (_Float16)w1;
            dst[b0 + 2] = (_Float16)w2;
            dst[b0 + 3] = (_Float16)w3;
        }
    };

    // ---- B-tile fill: 256 cols x 32 K halves, async global->LDS (32B/thread) ----
    auto fillB = [&](int buf, int t) {
        int col = tid >> 1;                            // 0..255
        int kh  = (tid & 1) * 16;                      // 0 or 16
        // Flat->LDS: low 32 bits of a generic pointer to __shared__ are the
        // workgroup-relative LDS byte offset (aperture layout), which is what
        // the async op's VDST expects.
        unsigned lds = (unsigned)(size_t)&sB[buf][col][kh];
        unsigned long long ga =
            (unsigned long long)(WcT + (size_t)(colBase + col) * K_TOT + t * BK + kh);
        asm volatile(
            "global_load_async_to_lds_b128 %0, %1, off\n\t"
            "global_load_async_to_lds_b128 %0, %1, off offset:16"
            :: "v"(lds), "v"(ga) : "memory");
    };

    fillA(0, 0);
    fillB(0, 0);

    for (int t = 0; t < KTILES; ++t) {
        // Drain this wave's async B copies, then rendezvous: after the barrier
        // both LDS buffers are consistent (compiler emits s_wait_dscnt 0 before
        // s_barrier_signal for the ds_store traffic).
        asm volatile("s_wait_asynccnt 0x0" ::: "memory");
        __syncthreads();
        const int cur = t & 1;
        if (t + 1 < KTILES) {          // prefetch next tile into the other buffer
            fillA(cur ^ 1, t + 1);
            fillB(cur ^ 1, t + 1);
        }

        // Preload ALL fragments, then issue the 8 WMMAs back-to-back.
        // A 16x32 frag: lane M=l16; 16 contiguous K starting at 16*half
        v16h afrag[2];
#pragma unroll
        for (int am = 0; am < 2; ++am) {
            int r = wm * 32 + am * 16 + l16;
            v8h lo = *(const v8h*)&sA[cur][r][16 * half];
            v8h hi = *(const v8h*)&sA[cur][r][16 * half + 8];
            afrag[am] = __builtin_shufflevector(lo, hi,
                0,1,2,3,4,5,6,7,8,9,10,11,12,13,14,15);
        }
        // B 32x16 frag: lane N=l16; 16 contiguous K starting at 16*half
        v16h bfrag[4];
#pragma unroll
        for (int bn = 0; bn < 4; ++bn) {
            int cc = wn * 64 + bn * 16 + l16;
            v8h lo = *(const v8h*)&sB[cur][cc][16 * half];
            v8h hi = *(const v8h*)&sB[cur][cc][16 * half + 8];
            bfrag[bn] = __builtin_shufflevector(lo, hi,
                0,1,2,3,4,5,6,7,8,9,10,11,12,13,14,15);
        }
#pragma unroll
        for (int bn = 0; bn < 4; ++bn)
#pragma unroll
            for (int am = 0; am < 2; ++am) {
                acc[am][bn] = __builtin_amdgcn_wmma_f32_16x16x32_f16(
                    false, afrag[am], false, bfrag[bn],
                    (short)0, acc[am][bn], false, false);
            }
    }

    // ---- epilogue: C layout -> lane l16 = N, VGPR r -> M = r + 8*half ----
#pragma unroll
    for (int am = 0; am < 2; ++am)
#pragma unroll
        for (int bn = 0; bn < 4; ++bn)
#pragma unroll
            for (int r = 0; r < 8; ++r) {
                int grow = rowBase + wm * 32 + am * 16 + r + 8 * half;
                int gcol = colBase + wn * 64 + bn * 16 + l16;
                out[(size_t)grow * OUT_F + gcol] = acc[am][bn][r];
            }
}

extern "C" void kernel_launch(void* const* d_in, const int* in_sizes, int n_in,
                              void* d_out, int out_size, void* d_ws, size_t ws_size,
                              hipStream_t stream) {
    const float* x  = (const float*)d_in[0];   // [16384, 512]
    const float* bw = (const float*)d_in[1];   // [512, 512]
    const float* sw = (const float*)d_in[2];   // [512, 512, 8]
    const float* ss = (const float*)d_in[3];   // [512, 512]
    float* out = (float*)d_out;                // [16384, 512]
    _Float16* WcT = (_Float16*)d_ws;           // [512, 4608] f16 = 4.5 MB

    int prep_blocks = (OUT_F * K_TOT + 255) / 256;
    kan_prep_w<<<prep_blocks, 256, 0, stream>>>(bw, sw, ss, WcT);

    dim3 grid(OUT_F / BN, BATCH / BM);         // (2, 128)
    kan_gemm<<<grid, NTHR, 0, stream>>>(x, WcT, out);
}
